// SoftKMeansClusterAssignment_71700184039814
// MI455X (gfx1250) — compile-verified, matrix operand order checked by reading, not run on hardware
//
#include <hip/hip_runtime.h>
#include <math.h>

// Soft K-Means cluster assignment for MI455X (gfx1250, wave32, WMMA).
// N=65536 samples, D=512 features, K=1024 classes, 10 iterations.
// Both GEMMs (f@c.T and a.T@f) run on v_wmma_f32_16x16x32_bf16 with f32
// accumulation; distances/softmax/normalization are done in f32.

#define N_SAMPLES 65536
#define DIM       512
#define K_CLASSES 1024
#define ITERS     10

typedef __attribute__((ext_vector_type(16))) __bf16 v16bf;
typedef __attribute__((ext_vector_type(8)))  float  v8f;

union FragU {
  v16bf v;
  unsigned short u[16];
  uint4 q[2];
};

__device__ __forceinline__ unsigned short f32_to_bf16(float f) {
  unsigned int x = __float_as_uint(f);
  x += 0x7fffu + ((x >> 16) & 1u);   // round to nearest even
  return (unsigned short)(x >> 16);
}

// Fragment where the contraction dim is CONTIGUOUS in memory per row.
// lane&15 selects the row (A: m, B: n); lanes 16-31 take the +8 K-offset.
// Per the CDNA5 16-bit A/B layout: regs 0-3 hold K = h..h+7, regs 4-7 hold
// K = 16+h..16+h+7 (h = 0 or 8 by lane half) -> two 16-byte loads.
__device__ __forceinline__ v16bf load_frag_rows(const unsigned short* base,
                                                int stride, int kk) {
  const int lane = threadIdx.x & 31;
  const unsigned short* p =
      base + (size_t)(lane & 15) * stride + kk + ((lane & 16) ? 8 : 0);
  FragU f;
  f.q[0] = *(const uint4*)(p);
  f.q[1] = *(const uint4*)(p + 16);
  return f.v;
}

// Fragment where the contraction dim runs ACROSS rows (stride `stride`);
// lane&15 selects the column. Used for B tiles staged row-major in LDS.
__device__ __forceinline__ v16bf load_frag_cols(const unsigned short* base,
                                                int stride, int col) {
  const int lane = threadIdx.x & 31;
  const int h = (lane & 16) ? 8 : 0;
  const unsigned short* p = base + col + (lane & 15);
  FragU f;
#pragma unroll
  for (int j = 0; j < 8; ++j) f.u[j] = p[(h + j) * stride];
#pragma unroll
  for (int j = 0; j < 8; ++j) f.u[8 + j] = p[(16 + h + j) * stride];
  return f.v;
}

// Convert rows of f32 data to bf16 and compute squared L2 norms.
// One wave per row; optional row gather through idx (for centroid init).
__global__ __launch_bounds__(256)
void k_rows_prep(const float* __restrict__ src, const int* __restrict__ idx,
                 unsigned short* __restrict__ dst, float* __restrict__ n2,
                 int rows) {
  const int wave = threadIdx.x >> 5;
  const int lane = threadIdx.x & 31;
  const int r = blockIdx.x * 8 + wave;
  if (r >= rows) return;
  const int sr = idx ? idx[r] : r;
  const float4* p = (const float4*)(src + (size_t)sr * DIM + lane * 16);
  FragU pk;
  float ss = 0.f;
#pragma unroll
  for (int q = 0; q < 4; ++q) {
    float4 v = p[q];
    ss += v.x * v.x + v.y * v.y + v.z * v.z + v.w * v.w;
    pk.u[q * 4 + 0] = f32_to_bf16(v.x);
    pk.u[q * 4 + 1] = f32_to_bf16(v.y);
    pk.u[q * 4 + 2] = f32_to_bf16(v.z);
    pk.u[q * 4 + 3] = f32_to_bf16(v.w);
  }
  uint4* d = (uint4*)(dst + (size_t)r * DIM + lane * 16);
  d[0] = pk.q[0];
  d[1] = pk.q[1];
#pragma unroll
  for (int m = 1; m < 32; m <<= 1) ss += __shfl_xor(ss, m, 32);
  if (lane == 0) n2[r] = ss;
}

__global__ void k_zero(float* __restrict__ p, int n) {
  int i = blockIdx.x * 256 + threadIdx.x;
  if (i < n) p[i] = 0.f;
}

// Fused: S = F_tile @ C^T (WMMA), d = sqrt(max(f2+c2-2S, eps)),
// softmax over all K=1024 classes, write assignments, column-sum atomics.
// Block: 256 threads (8 waves), 16 sample rows, each wave owns 128 classes.
__global__ __launch_bounds__(256)
void k_assign(const unsigned short* __restrict__ fb, const float* __restrict__ f2,
              const unsigned short* __restrict__ cb, const float* __restrict__ c2,
              float* __restrict__ out, float* __restrict__ cnt) {
  __shared__ __align__(16) unsigned short At[16 * 520];  // padded: conflict-free
  __shared__ float f2s[16];
  __shared__ float red[8][16];
  __shared__ float rowred[16];

  const int tid  = threadIdx.x;
  const int wave = tid >> 5;
  const int lane = tid & 31;
  const int row0 = blockIdx.x * 16;

  {  // stage 16x512 bf16 A tile (coalesced 64B per thread)
    const int r = tid >> 4;
    const int c = (tid & 15) * 32;
    const uint4* s = (const uint4*)(fb + (size_t)(row0 + r) * DIM + c);
    uint4* d = (uint4*)(At + r * 520 + c);
#pragma unroll
    for (int j = 0; j < 4; ++j) d[j] = s[j];
    if (tid < 16) f2s[tid] = f2[row0 + tid];
  }
  __syncthreads();

  v8f acc[8];
#pragma unroll
  for (int t = 0; t < 8; ++t)
#pragma unroll
    for (int v = 0; v < 8; ++v) acc[t][v] = 0.f;

  const int colw = wave * 128;
  for (int ks = 0; ks < DIM / 32; ++ks) {
    const int kk = ks * 32;
    v16bf aF = load_frag_rows(At, 520, kk);  // shared across the 8 col tiles
    // Issue ALL 16 B-fragment loads first (distinct regs) so they pipeline;
    // the sched_barrier keeps the scheduler from re-interleaving them with
    // the WMMAs (which would collapse them back onto one register pair and
    // serialize behind s_wait_loadcnt 0).  Loads of step ks+1 may still be
    // hoisted over the WMMAs of step ks -> software pipelining.
    v16bf bF[8];
#pragma unroll
    for (int t = 0; t < 8; ++t)
      bF[t] = load_frag_rows(cb + (size_t)(colw + t * 16) * DIM, DIM, kk);
    __builtin_amdgcn_sched_barrier(0);
#pragma unroll
    for (int t = 0; t < 8; ++t)
      acc[t] = __builtin_amdgcn_wmma_f32_16x16x32_bf16(
          false, aF, false, bF[t], (short)0, acc[t], false, false);
  }

  // ---- epilogue: distance -> softmax, all register-resident ----
  const int rb = (lane & 16) ? 8 : 0;  // row half per C/D layout
  float c2v[8];
#pragma unroll
  for (int t = 0; t < 8; ++t) c2v[t] = c2[colw + t * 16 + (lane & 15)];

  float pm[8];
#pragma unroll
  for (int v = 0; v < 8; ++v) pm[v] = -3.4e38f;
#pragma unroll
  for (int t = 0; t < 8; ++t)
#pragma unroll
    for (int v = 0; v < 8; ++v) {
      float d2 = f2s[rb + v] + c2v[t] - 2.f * acc[t][v];
      float lg = -sqrtf(fmaxf(d2, 1e-12f));  // logit = -dist / T, T = 1
      acc[t][v] = lg;
      pm[v] = fmaxf(pm[v], lg);
    }
  // row max: reduce the 16 lanes of each half, then across the 8 waves
#pragma unroll
  for (int m = 1; m < 16; m <<= 1)
#pragma unroll
    for (int v = 0; v < 8; ++v) pm[v] = fmaxf(pm[v], __shfl_xor(pm[v], m, 32));
  if ((lane & 15) == 0)
#pragma unroll
    for (int v = 0; v < 8; ++v) red[wave][rb + v] = pm[v];
  __syncthreads();
  if (tid < 16) {
    float m = red[0][tid];
#pragma unroll
    for (int w = 1; w < 8; ++w) m = fmaxf(m, red[w][tid]);
    rowred[tid] = m;
  }
  __syncthreads();
  float rmax[8];
#pragma unroll
  for (int v = 0; v < 8; ++v) rmax[v] = rowred[rb + v];

  float ps[8];
#pragma unroll
  for (int v = 0; v < 8; ++v) ps[v] = 0.f;
#pragma unroll
  for (int t = 0; t < 8; ++t)
#pragma unroll
    for (int v = 0; v < 8; ++v) {
      float e = __expf(acc[t][v] - rmax[v]);
      acc[t][v] = e;
      ps[v] += e;
    }
#pragma unroll
  for (int m = 1; m < 16; m <<= 1)
#pragma unroll
    for (int v = 0; v < 8; ++v) ps[v] += __shfl_xor(ps[v], m, 32);
  if ((lane & 15) == 0)
#pragma unroll
    for (int v = 0; v < 8; ++v) red[wave][rb + v] = ps[v];
  __syncthreads();
  if (tid < 16) {
    float s = red[0][tid];
#pragma unroll
    for (int w = 1; w < 8; ++w) s += red[w][tid];
    rowred[tid] = 1.f / s;
  }
  __syncthreads();
  float rinv[8];
#pragma unroll
  for (int v = 0; v < 8; ++v) rinv[v] = rowred[rb + v];

  float cs[8];
#pragma unroll
  for (int t = 0; t < 8; ++t) cs[t] = 0.f;
#pragma unroll
  for (int t = 0; t < 8; ++t)
#pragma unroll
    for (int v = 0; v < 8; ++v) {
      float a = acc[t][v] * rinv[v];
      out[(size_t)(row0 + rb + v) * K_CLASSES + colw + t * 16 + (lane & 15)] = a;
      cs[t] += a;
    }
  // column sums (weight_sum): fold both row halves, then one atomic per col
#pragma unroll
  for (int t = 0; t < 8; ++t) cs[t] += __shfl_xor(cs[t], 16, 32);
  if (lane < 16)
#pragma unroll
    for (int t = 0; t < 8; ++t) atomicAdd(&cnt[colw + t * 16 + lane], cs[t]);
}

// wsum[k][d] += sum_n a[n][k] * f[n][d]  (TN GEMM via LDS transpose of a).
// Grid: (K/16) x 16 N-chunks of 4096; f32 atomics merge the partials.
__global__ __launch_bounds__(256)
void k_update(const float* __restrict__ assign, const unsigned short* __restrict__ fb,
              float* __restrict__ wsum) {
  __shared__ __align__(16) unsigned short Al[16 * 40];   // a^T tile [k][n]
  __shared__ __align__(16) unsigned short Fl[32 * 520];  // f tile [n][d]

  const int tid  = threadIdx.x;
  const int wave = tid >> 5;
  const int lane = tid & 31;
  const int k0   = blockIdx.x * 16;
  const int nb0  = blockIdx.y * 4096;

  v8f acc[4];
#pragma unroll
  for (int t = 0; t < 4; ++t)
#pragma unroll
    for (int v = 0; v < 8; ++v) acc[t][v] = 0.f;

  for (int nb = nb0; nb < nb0 + 4096; nb += 32) {
    {  // stage a^T: 32n x 16k -> Al[k][n], f32 -> bf16 on the fly
      int i = tid;
#pragma unroll
      for (int rep = 0; rep < 2; ++rep) {
        const int nl = i >> 4;
        const int kl = i & 15;
        Al[kl * 40 + nl] =
            f32_to_bf16(assign[(size_t)(nb + nl) * K_CLASSES + k0 + kl]);
        i += 256;
      }
    }
    {  // stage features: 32n x 512d bf16, 128B per thread, coalesced
      const int r = tid >> 3;
      const int c = (tid & 7) * 64;
      const uint4* s = (const uint4*)(fb + (size_t)(nb + r) * DIM + c);
      uint4* d = (uint4*)(Fl + r * 520 + c);
#pragma unroll
      for (int j = 0; j < 8; ++j) d[j] = s[j];
    }
    __syncthreads();
    v16bf aF = load_frag_rows(Al, 40, 0);
    v16bf bF[4];
#pragma unroll
    for (int t = 0; t < 4; ++t)
      bF[t] = load_frag_cols(Fl, 520, wave * 64 + t * 16);
    __builtin_amdgcn_sched_barrier(0);
#pragma unroll
    for (int t = 0; t < 4; ++t)
      acc[t] = __builtin_amdgcn_wmma_f32_16x16x32_bf16(
          false, aF, false, bF[t], (short)0, acc[t], false, false);
    __syncthreads();
  }
  const int rb = (lane & 16) ? 8 : 0;
#pragma unroll
  for (int t = 0; t < 4; ++t)
#pragma unroll
    for (int v = 0; v < 8; ++v)
      atomicAdd(&wsum[(size_t)(k0 + rb + v) * DIM + wave * 64 + t * 16 + (lane & 15)],
                acc[t][v]);
}

// new_c = (wsum / cnt) row-L2-normalized; emit bf16 centroids + ||c||^2.
__global__ __launch_bounds__(256)
void k_normalize(const float* __restrict__ wsum, const float* __restrict__ cnt,
                 unsigned short* __restrict__ cb, float* __restrict__ c2) {
  const int wave = threadIdx.x >> 5;
  const int lane = threadIdx.x & 31;
  const int k = blockIdx.x * 8 + wave;
  const float inv = 1.f / cnt[k];  // softmax column sums are > 0
  const float4* p = (const float4*)(wsum + (size_t)k * DIM + lane * 16);
  float vals[16];
  float ss = 0.f;
#pragma unroll
  for (int q = 0; q < 4; ++q) {
    float4 x = p[q];
    vals[q * 4 + 0] = x.x * inv;
    vals[q * 4 + 1] = x.y * inv;
    vals[q * 4 + 2] = x.z * inv;
    vals[q * 4 + 3] = x.w * inv;
  }
#pragma unroll
  for (int j = 0; j < 16; ++j) ss += vals[j] * vals[j];
#pragma unroll
  for (int m = 1; m < 32; m <<= 1) ss += __shfl_xor(ss, m, 32);
  const float nrm = sqrtf(ss);
  const float den = fmaxf(nrm, 1e-12f);
  const float s = 1.f / den;
  FragU pk;
#pragma unroll
  for (int j = 0; j < 16; ++j) pk.u[j] = f32_to_bf16(vals[j] * s);
  uint4* d = (uint4*)(cb + (size_t)k * DIM + lane * 16);
  d[0] = pk.q[0];
  d[1] = pk.q[1];
  if (lane == 0) c2[k] = ss / (den * den);
}

extern "C" void kernel_launch(void* const* d_in, const int* in_sizes, int n_in,
                              void* d_out, int out_size, void* d_ws, size_t ws_size,
                              hipStream_t stream) {
  (void)in_sizes; (void)n_in; (void)out_size; (void)ws_size;
  const float* features = (const float*)d_in[0];
  const int*   init_idx = (const int*)d_in[1];
  float* out = (float*)d_out;

  // Workspace carve-out (~71 MB total, 256B-aligned slices).
  char* ws = (char*)d_ws;
  size_t o = 0;
  auto carve = [&](size_t bytes) -> char* {
    char* r = ws + o;
    o = (o + bytes + 255) & ~(size_t)255;
    return r;
  };
  unsigned short* f_bf16 = (unsigned short*)carve((size_t)N_SAMPLES * DIM * 2);
  float*          f2     = (float*)carve((size_t)N_SAMPLES * 4);
  unsigned short* c_bf16 = (unsigned short*)carve((size_t)K_CLASSES * DIM * 2);
  float*          c2     = (float*)carve((size_t)K_CLASSES * 4);
  float*          wsum   = (float*)carve((size_t)K_CLASSES * DIM * 4);
  float*          cnt    = (float*)carve((size_t)K_CLASSES * 4);

  // One-time prep: features -> bf16 + ||f||^2; gathered init centroids.
  k_rows_prep<<<N_SAMPLES / 8, 256, 0, stream>>>(features, nullptr, f_bf16, f2,
                                                 N_SAMPLES);
  k_rows_prep<<<K_CLASSES / 8, 256, 0, stream>>>(features, init_idx, c_bf16, c2,
                                                 K_CLASSES);

  for (int it = 0; it < ITERS; ++it) {
    k_zero<<<(K_CLASSES + 255) / 256, 256, 0, stream>>>(cnt, K_CLASSES);
    k_assign<<<N_SAMPLES / 16, 256, 0, stream>>>(f_bf16, f2, c_bf16, c2, out, cnt);
    if (it + 1 < ITERS) {  // last iteration's centroid update is unobservable
      k_zero<<<(K_CLASSES * DIM + 255) / 256, 256, 0, stream>>>(wsum,
                                                                K_CLASSES * DIM);
      k_update<<<dim3(K_CLASSES / 16, 16), 256, 0, stream>>>(out, f_bf16, wsum);
      k_normalize<<<K_CLASSES / 8, 256, 0, stream>>>(wsum, cnt, c_bf16, c2);
    }
  }
}
